// MODEL_1872605741740
// MI455X (gfx1250) — matrix operand order repaired and checked by reading, not compiled, not run on hardware
//
#include <hip/hip_runtime.h>
#include <stddef.h>

// ---------------------------------------------------------------------------
// Model constants (match the reference)
// ---------------------------------------------------------------------------
#define BB   2
#define TT   2048
#define CC   512
#define VV   32000
#define NN   16
#define HH   (CC / NN)       // 32 heads
#define MM   (BB * TT)       // 4096 rows
#define FF   (4 * CC)        // 2048 ffn hidden

// ---------------------------------------------------------------------------
// Types for WMMA
// ---------------------------------------------------------------------------
typedef __bf16 v16bf __attribute__((ext_vector_type(16)));
typedef float  v8f   __attribute__((ext_vector_type(8)));

union FragBF16 {
    uint4 q[2];
    v16bf v;
};

__device__ __forceinline__ float bf2f(unsigned short h) {
    unsigned u = ((unsigned)h) << 16;
    float f;
    __builtin_memcpy(&f, &u, 4);
    return f;
}
__device__ __forceinline__ unsigned short f2bf(float f) {
    unsigned u;
    __builtin_memcpy(&u, &f, 4);
    u += 0x7FFFu + ((u >> 16) & 1u);   // round-to-nearest-even
    return (unsigned short)(u >> 16);
}
__device__ __forceinline__ float sigmoidf(float x) { return 1.0f / (1.0f + __expf(-x)); }

// gfx1250 async global->LDS DMA (ASYNCcnt-tracked), 16B per lane.
// The LDS destination is passed as a *pointer* (not just an integer offset):
//  - low 32 bits of a shared-aperture generic address are the LDS byte offset
//    (ISA 10.2), which is what the VDST VGPR needs;
//  - passing the pointer into the asm captures the shared array, so the
//    compiler cannot assume LDS is never written and delete the ds_load
//    fragment reads (this happened when only the raw offset was passed).
__device__ __forceinline__ void async_copy16(void* lds_dst, const void* gsrc) {
    unsigned lds_off = (unsigned)(unsigned long long)lds_dst;
    asm volatile("global_load_async_to_lds_b128 %0, %1, off"
                 :: "v"(lds_off), "v"(gsrc), "v"(lds_dst)
                 : "memory");
}

// ---------------------------------------------------------------------------
// fp32 -> bf16 convert (weights & misc)
// ---------------------------------------------------------------------------
__global__ __launch_bounds__(256) void f32_to_bf16_kernel(
    const float* __restrict__ in, unsigned short* __restrict__ out, int n) {
    int i = blockIdx.x * 256 + threadIdx.x;
    if (i < n) out[i] = f2bf(in[i]);
}

// ---------------------------------------------------------------------------
// Embedding gather
// ---------------------------------------------------------------------------
__global__ __launch_bounds__(256) void embed_kernel(
    const int* __restrict__ tok, const float* __restrict__ emb,
    float* __restrict__ x) {
    int idx = blockIdx.x * 256 + threadIdx.x;     // over MM*CC
    int m = idx >> 9, c = idx & (CC - 1);
    x[idx] = emb[(size_t)tok[m] * CC + c];
}

// ---------------------------------------------------------------------------
// LayerNorm: one wave32 per row of 512. Optional f32 and/or bf16 outputs.
// ---------------------------------------------------------------------------
__global__ __launch_bounds__(256) void ln_kernel(
    const float* __restrict__ x, const float* __restrict__ g,
    const float* __restrict__ b, float* __restrict__ outf,
    unsigned short* __restrict__ outbf, int rows) {
    int wid  = (blockIdx.x * 256 + threadIdx.x) >> 5;
    int lane = threadIdx.x & 31;
    if (wid >= rows) return;
    const float* row = x + (size_t)wid * CC;
    float vals[16], s = 0.f, s2 = 0.f;
#pragma unroll
    for (int i = 0; i < 16; ++i) {
        float v = row[lane + i * 32];
        vals[i] = v; s += v; s2 += v * v;
    }
#pragma unroll
    for (int m = 16; m >= 1; m >>= 1) {
        s  += __shfl_xor(s, m, 32);
        s2 += __shfl_xor(s2, m, 32);
    }
    float mean = s * (1.0f / CC);
    float var  = s2 * (1.0f / CC) - mean * mean;
    float rstd = rsqrtf(var + 1e-5f);
#pragma unroll
    for (int i = 0; i < 16; ++i) {
        int c = lane + i * 32;
        float o = (vals[i] - mean) * rstd * g[c] + b[c];
        if (outf)  outf[(size_t)wid * CC + c] = o;
        if (outbf) outbf[(size_t)wid * CC + c] = f2bf(o);
    }
}

// ---------------------------------------------------------------------------
// tmix token-shift lerp -> 6 bf16 streams (xr xw xk xv xa xg)
// ---------------------------------------------------------------------------
__global__ __launch_bounds__(256) void lerp6_kernel(
    const float* __restrict__ xn,
    const float* __restrict__ cr, const float* __restrict__ cw,
    const float* __restrict__ ck, const float* __restrict__ cv,
    const float* __restrict__ ca, const float* __restrict__ cg,
    unsigned short* __restrict__ oxr, unsigned short* __restrict__ oxw,
    unsigned short* __restrict__ oxk, unsigned short* __restrict__ oxv,
    unsigned short* __restrict__ oxa, unsigned short* __restrict__ oxg) {
    int idx = blockIdx.x * 256 + threadIdx.x;   // over MM*CC
    int m = idx >> 9, c = idx & (CC - 1);
    int t = m & (TT - 1);
    float cur = xn[idx];
    float prv = (t > 0) ? xn[idx - CC] : 0.0f;
    float d = prv - cur;
    oxr[idx] = f2bf(cur + d * cr[c]);
    oxw[idx] = f2bf(cur + d * cw[c]);
    oxk[idx] = f2bf(cur + d * ck[c]);
    oxv[idx] = f2bf(cur + d * cv[c]);
    oxa[idx] = f2bf(cur + d * ca[c]);
    oxg[idx] = f2bf(cur + d * cg[c]);
}

// ---------------------------------------------------------------------------
// WMMA bf16 GEMM:  C[M,N] = A[M,K] * W[K,N] (+bias), f32 accumulate.
// Block = 256 thr = 8 waves.  Block tile 128x128, k-step 32.
// Wave tile 32(M) x 64(N) = 2x4 wmma accumulators.
// Double-buffered LDS tiles filled by GLOBAL_LOAD_ASYNC_TO_LDS_B128 DMA,
// pipelined one k-step ahead; in-order ASYNCcnt retirement lets
// `s_wait_asynccnt 4` cover the current tile while the next is in flight.
// LDS: A staged M-major (pad 40), W staged K-major (pad 136).
// M%128==0, N%128==0, K%32==0 (true for all uses here).
// ---------------------------------------------------------------------------
#define LDA   40
#define LDB   136
#define AELEM (128 * LDA)            // 5120 bf16
#define BELEM (32 * LDB)             // 4352 bf16
#define STAGE (AELEM + BELEM)        // 9472 bf16 per stage

__global__ __launch_bounds__(256) void gemm_bf16_kernel(
    const unsigned short* __restrict__ A,
    const unsigned short* __restrict__ W,
    float* __restrict__ C,
    const float* __restrict__ bias,
    int M, int N, int K) {
    __shared__ __align__(16) unsigned short SMEM[2 * STAGE];

    const int tid  = threadIdx.x;
    const int lane = tid & 31;
    const int wave = tid >> 5;
    const int wm   = wave >> 1;          // 0..3
    const int wn   = wave & 1;           // 0..1
    const int l15  = lane & 15;
    const int half = lane >> 4;
    const int blockM = blockIdx.y * 128;
    const int blockN = blockIdx.x * 128;

    v8f acc[2][4];
#pragma unroll
    for (int i = 0; i < 2; ++i)
#pragma unroll
        for (int j = 0; j < 4; ++j) acc[i][j] = 0.0f;

    // per-thread DMA coordinates
    const int arow = tid >> 2, a_k = (tid & 3) << 3;     // A tile: 128x32
    const int brow = tid >> 4, b_n = (tid & 15) << 3;    // W tile:  32x128

    // issue one 128x32 A tile + 32x128 W tile as 4 async 16B DMAs per thread
    auto issue_tile = [&](int k0, int buf) {
        unsigned short* base = SMEM + buf * STAGE;
#pragma unroll
        for (int c = 0; c < 2; ++c) {
            int row = arow + c * 64;
            async_copy16(base + row * LDA + a_k,
                         A + (size_t)(blockM + row) * K + k0 + a_k);
        }
#pragma unroll
        for (int c = 0; c < 2; ++c) {
            int row = brow + c * 16;
            async_copy16(base + AELEM + row * LDB + b_n,
                         W + (size_t)(k0 + row) * N + blockN + b_n);
        }
    };

    const int nk = K >> 5;
    issue_tile(0, 0);

    for (int it = 0; it < nk; ++it) {
        if (it + 1 < nk) {
            issue_tile((it + 1) << 5, (it + 1) & 1);
            // current tile (4 older DMAs) retired once <=4 remain outstanding
            asm volatile("s_wait_asynccnt 0x4" ::: "memory");
        } else {
            asm volatile("s_wait_asynccnt 0x0" ::: "memory");
        }
        __syncthreads();

        const unsigned short* As = SMEM + (it & 1) * STAGE;
        const unsigned short* Bs = As + AELEM;

        // ---- B fragments: lane holds N-col l15, K = 8*half+0..7 and 16+8*half+0..7
        FragBF16 bf[4];
        const unsigned short* bp = Bs + (half << 3) * LDB + wn * 64 + l15;
#pragma unroll
        for (int tj = 0; tj < 4; ++tj) {
#pragma unroll
            for (int c = 0; c < 2; ++c) {
                const unsigned short* p = bp + tj * 16 + c * 16 * LDB;
                bf[tj].q[c] = make_uint4(
                    (unsigned)p[0] | ((unsigned)p[LDB] << 16),
                    (unsigned)p[2 * LDB] | ((unsigned)p[3 * LDB] << 16),
                    (unsigned)p[4 * LDB] | ((unsigned)p[5 * LDB] << 16),
                    (unsigned)p[6 * LDB] | ((unsigned)p[7 * LDB] << 16));
            }
        }
        // ---- A fragments + WMMA
#pragma unroll
        for (int ti = 0; ti < 2; ++ti) {
            const unsigned short* ap = As + (wm * 32 + ti * 16 + l15) * LDA + (half << 3);
            FragBF16 af;
            af.q[0] = *(const uint4*)(ap);
            af.q[1] = *(const uint4*)(ap + 16);
#pragma unroll
            for (int tj = 0; tj < 4; ++tj) {
                acc[ti][tj] = __builtin_amdgcn_wmma_f32_16x16x32_bf16(
                    false, af.v, false, bf[tj].v, (short)0, acc[ti][tj],
                    false, false);
            }
        }
        __syncthreads();   // readers done before this stage is DMA'd again
    }

    // ---- epilogue: D layout — vgpr j -> M = j + 8*half, N = l15
#pragma unroll
    for (int ti = 0; ti < 2; ++ti) {
        int mbase = blockM + wm * 32 + ti * 16 + (half << 3);
#pragma unroll
        for (int tj = 0; tj < 4; ++tj) {
            int n = blockN + wn * 64 + tj * 16 + l15;
            float bv = bias ? bias[n] : 0.0f;
#pragma unroll
            for (int j = 0; j < 8; ++j)
                C[(size_t)(mbase + j) * N + n] = acc[ti][tj][j] + bv;
        }
    }
}

// ---------------------------------------------------------------------------
// LoRA down-projection: t[M,8] = act(x[M,512] @ W[512,8]); one wave per row.
// act: 0 = none, 1 = tanh, 2 = sigmoid
// ---------------------------------------------------------------------------
__global__ __launch_bounds__(256) void lora_down_kernel(
    const unsigned short* __restrict__ xbf, const float* __restrict__ W,
    float* __restrict__ t, int rows, int act) {
    int wid  = (blockIdx.x * 256 + threadIdx.x) >> 5;
    int lane = threadIdx.x & 31;
    if (wid >= rows) return;
    float acc[8];
#pragma unroll
    for (int j = 0; j < 8; ++j) acc[j] = 0.0f;
    const unsigned short* row = xbf + (size_t)wid * CC;
    for (int c = lane; c < CC; c += 32) {
        float xv = bf2f(row[c]);
#pragma unroll
        for (int j = 0; j < 8; ++j) acc[j] += xv * W[c * 8 + j];
    }
#pragma unroll
    for (int m = 16; m >= 1; m >>= 1)
#pragma unroll
        for (int j = 0; j < 8; ++j) acc[j] += __shfl_xor(acc[j], m, 32);
    if (lane == 0) {
#pragma unroll
        for (int j = 0; j < 8; ++j) {
            float v = acc[j];
            if (act == 1) v = tanhf(v);
            else if (act == 2) v = sigmoidf(v);
            t[(size_t)wid * 8 + j] = v;
        }
    }
}

// ---------------------------------------------------------------------------
// LoRA up-projection + fused epilogue.
// mode 0 (w):  out = -softplus(-(w0 + t@W2)) - 0.5
// mode 1 (a):  out = sigmoid(a0 + t@W2)
// mode 2 (v):  out = out + (vfirst - out) * sigmoid(v0 + t@W2)   [in-place v]
// mode 3 (g):  out = t@W2                                        [t pre-sigmoided]
// ---------------------------------------------------------------------------
__global__ __launch_bounds__(256) void lora_up_kernel(
    const float* __restrict__ t, const float* __restrict__ W2,
    const float* __restrict__ bias0, const float* __restrict__ vfirst,
    float* __restrict__ out, int mode) {
    int idx = blockIdx.x * 256 + threadIdx.x;     // over MM*CC
    int m = idx >> 9, c = idx & (CC - 1);
    float s = bias0 ? bias0[c] : 0.0f;
#pragma unroll
    for (int j = 0; j < 8; ++j) s += t[(size_t)m * 8 + j] * W2[j * CC + c];
    float val;
    if (mode == 0) {
        float z = -s;                                  // softplus(z), stable
        float sp = (z > 20.0f) ? z : log1pf(__expf(z));
        val = -sp - 0.5f;
    } else if (mode == 1) {
        val = sigmoidf(s);
    } else if (mode == 2) {
        float vv = out[idx];
        val = vv + (vfirst[idx] - vv) * sigmoidf(s);
    } else {
        val = s;
    }
    out[idx] = val;
}

// ---------------------------------------------------------------------------
// k preparation: kk = normalize_per_head(k*k_k); k_mod = k*(1+(a-1)*k_a)
// 16-element head reductions via shfl_xor inside wave32.
// ---------------------------------------------------------------------------
__global__ __launch_bounds__(256) void kprep_kernel(
    const float* __restrict__ k, const float* __restrict__ a_sig,
    const float* __restrict__ k_k, const float* __restrict__ k_a,
    float* __restrict__ kk, float* __restrict__ kmod) {
    int idx = blockIdx.x * 256 + threadIdx.x;     // over MM*CC
    int c = idx & (CC - 1);
    float kv = k[idx];
    float kkv = kv * k_k[c];
    float ss = kkv * kkv;
#pragma unroll
    for (int m = 8; m >= 1; m >>= 1) ss += __shfl_xor(ss, m, 32);
    float nrm = fmaxf(sqrtf(ss), 1e-12f);
    kk[idx]   = kkv / nrm;
    kmod[idx] = kv * (1.0f + (a_sig[idx] - 1.0f) * k_a[c]);
}

// ---------------------------------------------------------------------------
// RWKV-7 state recurrence.  One wave32 per (b,h) head-stream.
// Lane owns row i = lane&15 and 8 columns j in [8*(lane>>4), +8).
// ---------------------------------------------------------------------------
__global__ __launch_bounds__(32) void wkv7_kernel(
    const float* __restrict__ r, const float* __restrict__ w,
    const float* __restrict__ k, const float* __restrict__ v,
    const float* __restrict__ kk, const float* __restrict__ a,
    float* __restrict__ y) {
    int head = blockIdx.x;             // 0..63
    int b = head >> 5;                 // /HH
    int h = head & (HH - 1);
    int lane = threadIdx.x;
    int i = lane & 15;
    int half = lane >> 4;

    float S[8];
#pragma unroll
    for (int j = 0; j < 8; ++j) S[j] = 0.0f;

    size_t base = ((size_t)b * TT) * CC + h * NN;
    for (int t = 0; t < TT; ++t, base += CC) {
        float rv  = r[base + i];
        float wv  = w[base + i];
        float kv  = k[base + i];
        float vv  = v[base + i];   // lane already holds v[i]
        float kkv = kk[base + i];
        float av  = a[base + i];
        float dec = __expf(-__expf(wv));

        float dj[8], kj[8], kkj[8], aj[8], rj[8];
#pragma unroll
        for (int jj = 0; jj < 8; ++jj) {
            int j = half * 8 + jj;     // always 0..15 -> source lanes 0..15
            dj[jj]  = __shfl(dec, j, 32);
            kj[jj]  = __shfl(kv,  j, 32);
            kkj[jj] = __shfl(kkv, j, 32);
            aj[jj]  = __shfl(av,  j, 32);
            rj[jj]  = __shfl(rv,  j, 32);
        }
        // sa_i = sum_j S[i][j] * (-kk[j])
        float sa = 0.0f;
#pragma unroll
        for (int jj = 0; jj < 8; ++jj) sa -= S[jj] * kkj[jj];
        sa += __shfl_xor(sa, 16, 32);
        // S = S*diag(dec) + sa (x) b + v (x) k,  b = kk*a
#pragma unroll
        for (int jj = 0; jj < 8; ++jj) {
            float bj = kkj[jj] * aj[jj];
            S[jj] = S[jj] * dj[jj] + sa * bj + vv * kj[jj];
        }
        // y_i = sum_j S[i][j] * r[j]
        float yv = 0.0f;
#pragma unroll
        for (int jj = 0; jj < 8; ++jj) yv += S[jj] * rj[jj];
        yv += __shfl_xor(yv, 16, 32);
        if (half == 0) y[base + i] = yv;
    }
}

// ---------------------------------------------------------------------------
// Post-attention: groupnorm(y) + (sum_n r*k*r_k)*v, * g, -> bf16 for Wo GEMM
// ---------------------------------------------------------------------------
__global__ __launch_bounds__(256) void post_kernel(
    const float* __restrict__ y, const float* __restrict__ r,
    const float* __restrict__ kmod, const float* __restrict__ v,
    const float* __restrict__ g, const float* __restrict__ r_k,
    const float* __restrict__ gng, const float* __restrict__ gnb,
    unsigned short* __restrict__ out_bf) {
    int idx = blockIdx.x * 256 + threadIdx.x;     // over MM*CC
    int c = idx & (CC - 1);
    float yv = y[idx];
    float s = yv, s2 = yv * yv;
#pragma unroll
    for (int m = 8; m >= 1; m >>= 1) {
        s  += __shfl_xor(s, m, 32);
        s2 += __shfl_xor(s2, m, 32);
    }
    float mean = s * (1.0f / NN);
    float var  = s2 * (1.0f / NN) - mean * mean;
    float yn = (yv - mean) * rsqrtf(var + 64e-5f) * gng[c] + gnb[c];
    float rk = r[idx] * kmod[idx] * r_k[c];
#pragma unroll
    for (int m = 8; m >= 1; m >>= 1) rk += __shfl_xor(rk, m, 32);
    float o = yn + rk * v[idx];
    out_bf[idx] = f2bf(o * g[idx]);
}

// ---------------------------------------------------------------------------
// Residual add
// ---------------------------------------------------------------------------
__global__ __launch_bounds__(256) void add_kernel(
    float* __restrict__ x, const float* __restrict__ d) {
    int idx = blockIdx.x * 256 + threadIdx.x;
    x[idx] += d[idx];
}

// ---------------------------------------------------------------------------
// FFN token-shift lerp -> bf16
// ---------------------------------------------------------------------------
__global__ __launch_bounds__(256) void ffn_premix_kernel(
    const float* __restrict__ xn, const float* __restrict__ ck,
    unsigned short* __restrict__ out_bf) {
    int idx = blockIdx.x * 256 + threadIdx.x;     // over MM*CC
    int m = idx >> 9, c = idx & (CC - 1);
    int t = m & (TT - 1);
    float cur = xn[idx];
    float prv = (t > 0) ? xn[idx - CC] : 0.0f;
    out_bf[idx] = f2bf(cur + (prv - cur) * ck[c]);
}

// ---------------------------------------------------------------------------
// relu(h)^2 -> bf16
// ---------------------------------------------------------------------------
__global__ __launch_bounds__(256) void relusq_kernel(
    const float* __restrict__ h, unsigned short* __restrict__ out_bf, int n) {
    int idx = blockIdx.x * 256 + threadIdx.x;
    if (idx < n) {
        float v = fmaxf(h[idx], 0.0f);
        out_bf[idx] = f2bf(v * v);
    }
}

// ===========================================================================
// Host orchestration
// ===========================================================================
// Flat d_in layout (dict order of setup_inputs):
//  0 tokens ; 1 emb ; 2..11 ln1a_g/b ln1b_g/b ln2a_g/b ln2b_g/b lno_g/b
//  12..37 rwkv1 ; 38..63 rwkv2 ; 64..66 ffn1 ; 67..69 ffn2 ; 70 Wout ; 71 bout
// rwkv sub-offsets:
//  0 x_r 1 x_w 2 x_k 3 x_v 4 x_a 5 x_g 6 w0 7 w1 8 w2 9 a0 10 a1 11 a2
//  12 v0 13 v1 14 v2 15 g1 16 g2 17 k_k 18 k_a 19 r_k
//  20 Wr 21 Wk 22 Wv 23 Wo 24 gn_g 25 gn_b

extern "C" void kernel_launch(void* const* d_in, const int* in_sizes, int n_in,
                              void* d_out, int out_size, void* d_ws, size_t ws_size,
                              hipStream_t stream) {
    (void)in_sizes; (void)n_in; (void)out_size; (void)ws_size;

    const int MC  = MM * CC;           // 2,097,152
    const int MF  = MM * FF;           // 8,388,608
    const int GRID_MC = MC / 256;
    const int GRID_MF = MF / 256;
    const int GRID_ROWS = (MM * 32) / 256;   // wave-per-row kernels

    char* cur = (char*)d_ws;
    auto alloc = [&](size_t bytes) -> void* {
        void* p = (void*)cur;
        cur += (bytes + 255) & ~(size_t)255;
        return p;
    };

    // fp32 activations
    float* x     = (float*)alloc((size_t)MC * 4);
    float* xn    = (float*)alloc((size_t)MC * 4);
    float* rbuf  = (float*)alloc((size_t)MC * 4);
    float* kbuf  = (float*)alloc((size_t)MC * 4);
    float* v0buf = (float*)alloc((size_t)MC * 4);  // layer0 v == v_first
    float* v1buf = (float*)alloc((size_t)MC * 4);
    float* wraw  = (float*)alloc((size_t)MC * 4);
    float* asig  = (float*)alloc((size_t)MC * 4);
    float* kkbuf = (float*)alloc((size_t)MC * 4);
    float* kmod  = (float*)alloc((size_t)MC * 4);
    float* ybuf  = (float*)alloc((size_t)MC * 4);
    float* gbuf  = (float*)alloc((size_t)MC * 4);
    float* tmpo  = (float*)alloc((size_t)MC * 4);  // GEMM output / residual delta
    float* hbuf  = (float*)alloc((size_t)MF * 4);
    float* tsm   = (float*)alloc((size_t)MM * 8 * 4);
    // bf16 activations
    unsigned short* xr_bf = (unsigned short*)alloc((size_t)MC * 2);
    unsigned short* xw_bf = (unsigned short*)alloc((size_t)MC * 2);
    unsigned short* xk_bf = (unsigned short*)alloc((size_t)MC * 2);
    unsigned short* xv_bf = (unsigned short*)alloc((size_t)MC * 2);
    unsigned short* xa_bf = (unsigned short*)alloc((size_t)MC * 2);
    unsigned short* xg_bf = (unsigned short*)alloc((size_t)MC * 2);
    unsigned short* gy_bf = (unsigned short*)alloc((size_t)MC * 2);
    unsigned short* hx_bf = (unsigned short*)alloc((size_t)MC * 2);
    unsigned short* h_bf  = (unsigned short*)alloc((size_t)MF * 2);
    unsigned short* fin_bf= (unsigned short*)alloc((size_t)MC * 2);
    // bf16 weights
    unsigned short* Wr_bf[2], *Wk_bf[2], *Wv_bf[2], *Wo_bf[2];
    unsigned short* Fk_bf[2], *Fv_bf[2];
    for (int l = 0; l < 2; ++l) {
        Wr_bf[l] = (unsigned short*)alloc((size_t)CC * CC * 2);
        Wk_bf[l] = (unsigned short*)alloc((size_t)CC * CC * 2);
        Wv_bf[l] = (unsigned short*)alloc((size_t)CC * CC * 2);
        Wo_bf[l] = (unsigned short*)alloc((size_t)CC * CC * 2);
        Fk_bf[l] = (unsigned short*)alloc((size_t)CC * FF * 2);
        Fv_bf[l] = (unsigned short*)alloc((size_t)FF * CC * 2);
    }
    unsigned short* Wout_bf = (unsigned short*)alloc((size_t)CC * VV * 2);

    auto F = [&](int i) { return (const float*)d_in[i]; };

    auto cvt = [&](int idx, unsigned short* dst, int n) {
        f32_to_bf16_kernel<<<(n + 255) / 256, 256, 0, stream>>>(F(idx), dst, n);
    };
    auto gemm = [&](const unsigned short* A, const unsigned short* W, float* C,
                    int M_, int N_, int K_, const float* bias) {
        dim3 grid(N_ / 128, M_ / 128);
        gemm_bf16_kernel<<<grid, 256, 0, stream>>>(A, W, C, bias, M_, N_, K_);
    };

    // ---- weight conversion (deterministic, every call) ----
    for (int l = 0; l < 2; ++l) {
        int rb = l ? 38 : 12;
        int fb = l ? 67 : 64;
        cvt(rb + 20, Wr_bf[l], CC * CC);
        cvt(rb + 21, Wk_bf[l], CC * CC);
        cvt(rb + 22, Wv_bf[l], CC * CC);
        cvt(rb + 23, Wo_bf[l], CC * CC);
        cvt(fb + 1,  Fk_bf[l], CC * FF);
        cvt(fb + 2,  Fv_bf[l], FF * CC);
    }
    cvt(70, Wout_bf, CC * VV);

    // ---- embedding ----
    embed_kernel<<<GRID_MC, 256, 0, stream>>>((const int*)d_in[0], F(1), x);

    // ---- two transformer blocks ----
    for (int l = 0; l < 2; ++l) {
        int rb = l ? 38 : 12;
        int fb = l ? 67 : 64;
        const float* lnAg = F(l ? 6 : 2); const float* lnAb = F(l ? 7 : 3);
        const float* lnBg = F(l ? 8 : 4); const float* lnBb = F(l ? 9 : 5);
        float* vbuf = l ? v1buf : v0buf;

        // --- attention (tmix) ---
        ln_kernel<<<GRID_ROWS, 256, 0, stream>>>(x, lnAg, lnAb, xn, nullptr, MM);
        lerp6_kernel<<<GRID_MC, 256, 0, stream>>>(
            xn, F(rb + 0), F(rb + 1), F(rb + 2), F(rb + 3), F(rb + 4), F(rb + 5),
            xr_bf, xw_bf, xk_bf, xv_bf, xa_bf, xg_bf);

        gemm(xr_bf, Wr_bf[l], rbuf, MM, CC, CC, nullptr);
        gemm(xk_bf, Wk_bf[l], kbuf, MM, CC, CC, nullptr);
        gemm(xv_bf, Wv_bf[l], vbuf, MM, CC, CC, nullptr);

        // w LoRA: tanh down, softplus up
        lora_down_kernel<<<GRID_ROWS, 256, 0, stream>>>(xw_bf, F(rb + 7), tsm, MM, 1);
        lora_up_kernel<<<GRID_MC, 256, 0, stream>>>(tsm, F(rb + 8), F(rb + 6), nullptr, wraw, 0);
        // a LoRA
        lora_down_kernel<<<GRID_ROWS, 256, 0, stream>>>(xa_bf, F(rb + 10), tsm, MM, 0);
        lora_up_kernel<<<GRID_MC, 256, 0, stream>>>(tsm, F(rb + 11), F(rb + 9), nullptr, asig, 1);
        // v mixing (layer 1 only)
        if (l == 1) {
            lora_down_kernel<<<GRID_ROWS, 256, 0, stream>>>(xv_bf, F(rb + 13), tsm, MM, 0);
            lora_up_kernel<<<GRID_MC, 256, 0, stream>>>(tsm, F(rb + 14), F(rb + 12), v0buf, vbuf, 2);
        }
        // g LoRA: sigmoid down, plain up
        lora_down_kernel<<<GRID_ROWS, 256, 0, stream>>>(xg_bf, F(rb + 15), tsm, MM, 2);
        lora_up_kernel<<<GRID_MC, 256, 0, stream>>>(tsm, F(rb + 16), nullptr, nullptr, gbuf, 3);

        kprep_kernel<<<GRID_MC, 256, 0, stream>>>(kbuf, asig, F(rb + 17), F(rb + 18), kkbuf, kmod);
        wkv7_kernel<<<BB * HH, 32, 0, stream>>>(rbuf, wraw, kmod, vbuf, kkbuf, asig, ybuf);
        post_kernel<<<GRID_MC, 256, 0, stream>>>(
            ybuf, rbuf, kmod, vbuf, gbuf, F(rb + 19), F(rb + 24), F(rb + 25), gy_bf);
        gemm(gy_bf, Wo_bf[l], tmpo, MM, CC, CC, nullptr);
        add_kernel<<<GRID_MC, 256, 0, stream>>>(x, tmpo);

        // --- FFN ---
        ln_kernel<<<GRID_ROWS, 256, 0, stream>>>(x, lnBg, lnBb, xn, nullptr, MM);
        ffn_premix_kernel<<<GRID_MC, 256, 0, stream>>>(xn, F(fb + 0), hx_bf);
        gemm(hx_bf, Fk_bf[l], hbuf, MM, FF, CC, nullptr);
        relusq_kernel<<<GRID_MF, 256, 0, stream>>>(hbuf, h_bf, MF);
        gemm(h_bf, Fv_bf[l], tmpo, MM, CC, FF, nullptr);
        add_kernel<<<GRID_MC, 256, 0, stream>>>(x, tmpo);
    }

    // ---- final LN + LM head ----
    ln_kernel<<<GRID_ROWS, 256, 0, stream>>>(x, F(10), F(11), nullptr, fin_bf, MM);
    gemm(fin_bf, Wout_bf, (float*)d_out, MM, VV, CC, F(71));
}
